// MISSolver_8942121910956
// MI455X (gfx1250) — compile-verified, matrix operand order checked by reading, not run on hardware
//
#include <hip/hip_runtime.h>
#include <hip/hip_bf16.h>

// ---------------------------------------------------------------------------
// MIS GNN solver for MI455X (gfx1250, wave32).
//  * 13 (32768x64)@(64x64) MLP layers -> v_wmma_f32_16x16x32_f16 (f32 accum)
//  * adjacency as bitmask (2MB, L2-resident) for threshold + greedy phases
//  * greedy phase: one workgroup per graph, bitonic sort + LDS reductions,
//    global_prefetch_b8 for next adjacency row.
// Assumed d_in flattening (jax pytree: top-level insertion order, nested
// dicts sorted alphabetically, mlp lists in order, tuples (W,b)):
//  0:w | 1,2:bn1 beta,gamma | 3,4:bn2 | 5,6:bn3 | 7,8:bn5
//  9:conv1.eps 10..15:conv1 (W0 1x64,b0,W1,b1,W2,b2)
//  16..22:conv2 | 23..29:conv3 | 30..36:conv5 | 37..43:conv6 (W2 64x1,b2 1)
//  44:edge_index (2 x 1048576 i32) | 45:batch (unused)
// ---------------------------------------------------------------------------

#define NB 64          // graphs
#define NN 512         // nodes per graph
#define NV (NB*NN)     // 32768 rows
#define NC 64          // channels
#define NE 1048576     // directed edges (both orientations)
#define NWORD 16       // 512 bits / 32

typedef __attribute__((ext_vector_type(16))) _Float16 v16h;
typedef __attribute__((ext_vector_type(8)))  float    v8f;

// ------------------------------- utility -----------------------------------
__global__ void k_zero_u32(unsigned* p, int n) {
    int i = blockIdx.x * blockDim.x + threadIdx.x;
    if (i < n) p[i] = 0u;
}

// out[i] = (1 + *eps) * x[i]
__global__ void k_scale(const float* __restrict__ x, const float* __restrict__ eps,
                        float* __restrict__ out, int n) {
    int i = blockIdx.x * blockDim.x + threadIdx.x;
    if (i < n) out[i] = (1.0f + eps[0]) * x[i];
}

// ------------------------------ adjacency ----------------------------------
__global__ void k_adj_build(const int* __restrict__ s, const int* __restrict__ d,
                            unsigned* __restrict__ adj, int nE) {
    int e = blockIdx.x * blockDim.x + threadIdx.x;
    if (e >= nE) return;
    int si = s[e], di = d[e];
    int b = si / NN, row = si % NN, col = di % NN;
    atomicOr(adj + (b * NN + row) * NWORD + (col >> 5), 1u << (col & 31));
}

// ------------------------------ GIN scatter --------------------------------
__global__ void k_edge_agg1(const int* __restrict__ s, const int* __restrict__ d,
                            const float* __restrict__ x, float* __restrict__ agg, int nE) {
    int e = blockIdx.x * blockDim.x + threadIdx.x;
    if (e >= nE) return;
    atomicAdd(agg + d[e], x[s[e]]);
}

__global__ void k_edge_agg64(const int* __restrict__ s, const int* __restrict__ d,
                             const float* __restrict__ x, float* __restrict__ agg, int nE) {
    int e = blockIdx.x * blockDim.x + threadIdx.x;
    if (e >= nE) return;
    const float* xr = x + (long)s[e] * NC;
    float* ar = agg + (long)d[e] * NC;
    #pragma unroll 8
    for (int c = 0; c < NC; ++c) atomicAdd(ar + c, xr[c]);
}

// -------------------------- weight pre-swizzle -----------------------------
// frag[((kt*4+nt)*32+lane)*16 + e] holds the B fragment (K x N, row major W)
// in the CDNA5 16-bit 16x32 lane layout (CDNA5 ISA 7.12.2).
__global__ void k_prep_wfrag(const float* __restrict__ W, _Float16* __restrict__ frag) {
    int id = threadIdx.x;                 // 256 = 2kt * 4nt * 32lane
    if (id >= 256) return;
    int lane = id & 31, nt = (id >> 5) & 3, kt = id >> 7;
    int n  = nt * 16 + (lane & 15);
    int kb = ((lane >> 4) << 3) + kt * 32;
    _Float16* dst = frag + ((kt * 4 + nt) * 32 + lane) * 16;
    #pragma unroll
    for (int e = 0; e < 8; ++e) dst[e]     = (_Float16)W[(kb + e) * NC + n];
    #pragma unroll
    for (int e = 0; e < 8; ++e) dst[8 + e] = (_Float16)W[(kb + 16 + e) * NC + n];
}

// ------------------------------- WMMA GEMM ---------------------------------
// Y[32768,64] = act(X[32768,64] @ W + bias). One wave -> one 16-row strip.
__global__ __launch_bounds__(128) void
k_gemm64(const float* __restrict__ X, const _Float16* __restrict__ frag,
         const float* __restrict__ bias, float* __restrict__ Y, int doRelu) {
    const int lane  = threadIdx.x & 31;
    const int mtile = blockIdx.x * (blockDim.x >> 5) + (threadIdx.x >> 5);   // 0..2047
    const int row   = mtile * 16 + (lane & 15);
    const int kb    = (lane >> 4) << 3;

    v8f acc[4] = {};
    #pragma unroll
    for (int kt = 0; kt < 2; ++kt) {
        const float* xr = X + (long)row * NC + kt * 32 + kb;
        v16h a;
        #pragma unroll
        for (int e = 0; e < 8; ++e) a[e]     = (_Float16)xr[e];
        #pragma unroll
        for (int e = 0; e < 8; ++e) a[8 + e] = (_Float16)xr[16 + e];
        #pragma unroll
        for (int nt = 0; nt < 4; ++nt) {
            v16h b = *(const v16h*)(frag + ((kt * 4 + nt) * 32 + lane) * 16);
            acc[nt] = __builtin_amdgcn_wmma_f32_16x16x32_f16(
                false, a, false, b, (short)0, acc[nt], false, false);
        }
    }
    const int mrow = mtile * 16 + ((lane >> 4) << 3);
    #pragma unroll
    for (int nt = 0; nt < 4; ++nt) {
        int n = nt * 16 + (lane & 15);
        float bv = bias[n];
        #pragma unroll
        for (int r = 0; r < 8; ++r) {
            float v = acc[nt][r] + bv;
            if (doRelu) v = v > 0.f ? v : 0.f;
            Y[(long)(mrow + r) * NC + n] = v;
        }
    }
}

// --------------------------- small MLP layers ------------------------------
// conv1 layer0: 1 -> 64
__global__ void k_expand1(const float* __restrict__ x1, const float* __restrict__ W0,
                          const float* __restrict__ b0, float* __restrict__ Y) {
    int i = blockIdx.x * blockDim.x + threadIdx.x;   // over 32768*64
    if (i >= NV * NC) return;
    int n = i & 63, r = i >> 6;
    float v = x1[r] * W0[n] + b0[n];
    Y[i] = v > 0.f ? v : 0.f;
}

// conv6 layer2: 64 -> 1, + sigmoid
__global__ void k_dot_sig(const float* __restrict__ X, const float* __restrict__ W,
                          const float* __restrict__ b, float* __restrict__ prob) {
    int i = blockIdx.x * blockDim.x + threadIdx.x;
    if (i >= NV) return;
    float s = b[0];
    const float* xr = X + (long)i * NC;
    #pragma unroll 8
    for (int k = 0; k < NC; ++k) s += xr[k] * W[k];
    prob[i] = 1.f / (1.f + __expf(-s));
}

// ------------------------------ BatchNorm ----------------------------------
__global__ __launch_bounds__(256) void
k_bn_stats(const float* __restrict__ Y, float* __restrict__ stats) {
    __shared__ float s0[256], s1[256];
    int t = threadIdx.x, c = t & 63, rl = t >> 6;            // 4 row-lanes
    float sum = 0.f, sq = 0.f;
    for (int i = blockIdx.x * 4 + rl; i < NV; i += gridDim.x * 4) {
        float v = Y[(long)i * NC + c];
        sum += v; sq += v * v;
    }
    s0[t] = sum; s1[t] = sq;
    __syncthreads();
    if (t < 64) {
        float a = s0[t] + s0[t + 64] + s0[t + 128] + s0[t + 192];
        float b = s1[t] + s1[t + 64] + s1[t + 128] + s1[t + 192];
        atomicAdd(stats + c, a);
        atomicAdd(stats + 64 + c, b);
    }
}

__global__ void k_bn_apply(const float* __restrict__ Y, const float* __restrict__ stats,
                           const float* __restrict__ gamma, const float* __restrict__ beta,
                           float* __restrict__ out) {
    int i = blockIdx.x * blockDim.x + threadIdx.x;
    if (i >= NV * NC) return;
    int c = i & 63;
    float m = stats[c] * (1.f / NV);
    float v = stats[64 + c] * (1.f / NV) - m * m;
    float r = __frsqrt_rn(v + 1e-5f);
    float y = gamma[c] * (Y[i] - m) * r + beta[c];
    out[i] = y > 0.f ? y : 0.f;                  // every BN is followed by ReLU
}

// --------------------------- thresholds + loss -----------------------------
__global__ __launch_bounds__(512) void
k_threshold(const float* __restrict__ w, const float* __restrict__ prob,
            const unsigned* __restrict__ adj, float* __restrict__ thr,
            float* __restrict__ gam, float* __restrict__ loss) {
    __shared__ float sp[NN], r0[NN], r1[NN], r2[NN];
    int b = blockIdx.x, t = threadIdx.x, base = b * NN;
    float pv = prob[base + t], wv = w[base + t];
    sp[t] = pv;
    __syncthreads();
    float y = 0.f;
    const unsigned* row = adj + (long)(base + t) * NWORD;
    #pragma unroll
    for (int q = 0; q < NWORD; ++q) {
        unsigned m = row[q]; int o = q * 32;
        while (m) { int j = __builtin_ctz(m); m &= m - 1; y += sp[o + j]; }
    }
    r0[t] = wv; r1[t] = wv * pv; r2[t] = pv * y;
    __syncthreads();
    for (int s = 256; s > 0; s >>= 1) {
        if (t < s) { r0[t] += r0[t + s]; r1[t] += r1[t + s]; r2[t] += r2[t + s]; }
        __syncthreads();
    }
    if (t == 0) {
        float g = r0[0], tb = g - r1[0] + r2[0];
        gam[b] = g; thr[b] = tb;
        atomicAdd(loss, tb * (1.f / NB));
    }
}

// ------------------------------- greedy ------------------------------------
__global__ __launch_bounds__(512) void
k_greedy(const float* __restrict__ w, const float* __restrict__ prob,
         const unsigned* __restrict__ adj, const float* __restrict__ thr,
         const float* __restrict__ gam, float* __restrict__ outSel) {
    __shared__ float xs[NN], tp[NN], r0[NN], r1[NN], skey[NN];
    __shared__ int   sord[NN];
    __shared__ unsigned char ssel[NN], srej[NN];
    __shared__ int sHas;

    int b = blockIdx.x, t = threadIdx.x, base = b * NN;
    float wv = w[base + t];
    float thrb = thr[b], gb = gam[b];
    xs[t] = prob[base + t];
    skey[t] = xs[t]; sord[t] = t;
    ssel[t] = 0; srej[t] = 0;
    __syncthreads();

    // bitonic sort: descending by prob, ties by index ascending (argsort(-x))
    for (int k = 2; k <= NN; k <<= 1) {
        for (int j = k >> 1; j > 0; j >>= 1) {
            __syncthreads();
            int p = t ^ j;
            if (p > t) {
                float ka = skey[t], kb2 = skey[p];
                int ia = sord[t], ib = sord[p];
                bool pBefore = (kb2 > ka) || (kb2 == ka && ib < ia);
                bool dirDesc = ((t & k) == 0);
                if (pBefore == dirDesc) {
                    skey[t] = kb2; skey[p] = ka;
                    sord[t] = ib;  sord[p] = ia;
                }
            }
        }
    }
    __syncthreads();

    const unsigned* myrow = adj + (long)(base + t) * NWORD;
    for (int step = 0; step < NN; ++step) {
        if (t == 0) sHas = 0;
        __syncthreads();
        int node = sord[step];
        if (step + 1 < NN && t < NWORD)   // pull next row toward L2/L0 early
            __builtin_prefetch(adj + (long)(base + sord[step + 1]) * NWORD + t, 0, 0);
        unsigned nw = adj[(long)(base + node) * NWORD + (t >> 5)];
        bool neigh = (nw >> (t & 31)) & 1u;
        if (neigh) atomicOr(&sHas, 1);
        float tv = neigh ? 0.f : xs[t];
        if (t == node) tv = 1.f;
        tp[t] = tv;
        __syncthreads();

        float y = 0.f;
        #pragma unroll
        for (int q = 0; q < NWORD; ++q) {
            unsigned m = myrow[q]; int o = q * 32;
            while (m) { int j = __builtin_ctz(m); m &= m - 1; y += tp[o + j]; }
        }
        r0[t] = wv * tv; r1[t] = y * tv;
        __syncthreads();
        for (int s = 256; s > 0; s >>= 1) {
            if (t < s) { r0[t] += r0[t + s]; r1[t] += r1[t + s]; }
            __syncthreads();
        }
        float l = gb - r0[0] + r1[0];
        bool has = sHas != 0;
        bool accept = has && !srej[node] && !ssel[node] && (l <= thrb);
        __syncthreads();
        if (accept) { xs[t] = tv; if (neigh) srej[t] = 1; }
        if (t == 0) ssel[node] = (unsigned char)(ssel[node] | accept | (!has));
        __syncthreads();
    }
    outSel[base + t] = ssel[t] ? 1.f : 0.f;
}

// ------------------------------ launch glue --------------------------------
extern "C" void kernel_launch(void* const* d_in, const int* in_sizes, int n_in,
                              void* d_out, int out_size, void* d_ws, size_t ws_size,
                              hipStream_t stream) {
    const float* w = (const float*)d_in[0];
    const float* bn_beta[4]  = {(const float*)d_in[1], (const float*)d_in[3],
                                (const float*)d_in[5], (const float*)d_in[7]};
    const float* bn_gamma[4] = {(const float*)d_in[2], (const float*)d_in[4],
                                (const float*)d_in[6], (const float*)d_in[8]};
    // conv param blocks: eps, W0,b0, W1,b1, W2,b2 at 9,16,23,30,37
    const int cb[5] = {9, 16, 23, 30, 37};
    const int* src = (const int*)d_in[44];
    const int* dst = src + NE;

    // workspace layout
    char* p = (char*)d_ws;
    float* bufA = (float*)p;  p += (size_t)NV * NC * 4;       // 8 MB
    float* bufB = (float*)p;  p += (size_t)NV * NC * 4;       // 8 MB
    float* bufC = (float*)p;  p += (size_t)NV * NC * 4;       // 8 MB
    float* prob = (float*)p;  p += (size_t)NV * 4;            // 128 KB
    unsigned* adj = (unsigned*)p; p += (size_t)NV * NWORD * 4;// 2 MB
    _Float16* wfrag = (_Float16*)p; p += 13 * 4096 * 2;       // 104 KB
    float* stats = (float*)p; p += 512;                       // 128 f32 (+pad)
    float* thr = (float*)p;   p += 256;
    float* gam = (float*)p;   p += 256;

    float* loss = (float*)d_out;
    float* sel  = (float*)d_out + 1;

    // adjacency bitmask
    k_zero_u32<<<(NV * NWORD + 255) / 256, 256, 0, stream>>>(adj, NV * NWORD);
    k_adj_build<<<(NE + 255) / 256, 256, 0, stream>>>(src, dst, adj, NE);

    // pre-swizzle 13 fat (64x64) weights into WMMA B-fragment layout
    const int fatW[13] = {cb[0]+3, cb[0]+5,                    // conv1.W1, W2
                          cb[1]+1, cb[1]+3, cb[1]+5,           // conv2.W0..W2
                          cb[2]+1, cb[2]+3, cb[2]+5,           // conv3
                          cb[3]+1, cb[3]+3, cb[3]+5,           // conv5
                          cb[4]+1, cb[4]+3};                   // conv6.W0, W1
    for (int l = 0; l < 13; ++l)
        k_prep_wfrag<<<1, 256, 0, stream>>>((const float*)d_in[fatW[l]],
                                            wfrag + (size_t)l * 4096);

    const int G2M = (NV * NC + 255) / 256;   // element-wise grids
    const int GGE = 512;                     // 2048 mtiles / 4 waves per block

    // ---- conv1: x = w[:,None] ----
    k_scale<<<(NV + 255) / 256, 256, 0, stream>>>(w, (const float*)d_in[cb[0]], bufB, NV);
    k_edge_agg1<<<(NE + 255) / 256, 256, 0, stream>>>(src, dst, w, bufB, NE);
    k_expand1<<<G2M, 256, 0, stream>>>(bufB, (const float*)d_in[cb[0]+1],
                                       (const float*)d_in[cb[0]+2], bufC);
    k_gemm64<<<GGE, 128, 0, stream>>>(bufC, wfrag + 0*4096, (const float*)d_in[cb[0]+4], bufB, 1);
    k_gemm64<<<GGE, 128, 0, stream>>>(bufB, wfrag + 1*4096, (const float*)d_in[cb[0]+6], bufC, 0);
    k_zero_u32<<<1, 128, 0, stream>>>((unsigned*)stats, 128);
    k_bn_stats<<<256, 256, 0, stream>>>(bufC, stats);
    k_bn_apply<<<G2M, 256, 0, stream>>>(bufC, stats, bn_gamma[0], bn_beta[0], bufA);

    // ---- conv2/3/5 ----
    for (int c = 1; c <= 3; ++c) {
        int e = cb[c], f = 2 + (c - 1) * 3;
        k_scale<<<G2M, 256, 0, stream>>>(bufA, (const float*)d_in[e], bufB, NV * NC);
        k_edge_agg64<<<(NE + 255) / 256, 256, 0, stream>>>(src, dst, bufA, bufB, NE);
        k_gemm64<<<GGE, 128, 0, stream>>>(bufB, wfrag + (size_t)(f+0)*4096, (const float*)d_in[e+2], bufC, 1);
        k_gemm64<<<GGE, 128, 0, stream>>>(bufC, wfrag + (size_t)(f+1)*4096, (const float*)d_in[e+4], bufB, 1);
        k_gemm64<<<GGE, 128, 0, stream>>>(bufB, wfrag + (size_t)(f+2)*4096, (const float*)d_in[e+6], bufC, 0);
        k_zero_u32<<<1, 128, 0, stream>>>((unsigned*)stats, 128);
        k_bn_stats<<<256, 256, 0, stream>>>(bufC, stats);
        k_bn_apply<<<G2M, 256, 0, stream>>>(bufC, stats, bn_gamma[c], bn_beta[c], bufA);
    }

    // ---- conv6 -> prob ----
    {
        int e = cb[4];
        k_scale<<<G2M, 256, 0, stream>>>(bufA, (const float*)d_in[e], bufB, NV * NC);
        k_edge_agg64<<<(NE + 255) / 256, 256, 0, stream>>>(src, dst, bufA, bufB, NE);
        k_gemm64<<<GGE, 128, 0, stream>>>(bufB, wfrag + 11*4096, (const float*)d_in[e+2], bufC, 1);
        k_gemm64<<<GGE, 128, 0, stream>>>(bufC, wfrag + 12*4096, (const float*)d_in[e+4], bufB, 1);
        k_dot_sig<<<(NV + 255) / 256, 256, 0, stream>>>(bufB, (const float*)d_in[e+5],
                                                        (const float*)d_in[e+6], prob);
    }

    // ---- thresholds, loss, greedy ----
    k_zero_u32<<<1, 32, 0, stream>>>((unsigned*)loss, 1);
    k_threshold<<<NB, NN, 0, stream>>>(w, prob, adj, thr, gam, loss);
    k_greedy<<<NB, NN, 0, stream>>>(w, prob, adj, thr, gam, sel);
}
